// MultiHeadSelfAttention_76974403879612
// MI455X (gfx1250) — compile-verified, matrix operand order checked by reading, not run on hardware
//
#include <hip/hip_runtime.h>

// ---------------------------------------------------------------------------
// Types
// ---------------------------------------------------------------------------
typedef __bf16 bf16;
typedef __attribute__((ext_vector_type(16))) __bf16 bf16x16;
typedef __attribute__((ext_vector_type(8)))  __bf16 bf16x8;
typedef __attribute__((ext_vector_type(4)))  __bf16 bf16x4;
typedef __attribute__((ext_vector_type(8)))  float  f32x8;
typedef __attribute__((ext_vector_type(4)))  float  f32x4;

union Frag {
    bf16x16 v;
    struct { bf16x8 lo, hi; } h;
};

__device__ __forceinline__ f32x8 wmma_bf16(Frag a, Frag b, f32x8 c) {
    // D(16x16 f32) = A(16x32 bf16) * B(32x16 bf16) + C
    return __builtin_amdgcn_wmma_f32_16x16x32_bf16(
        /*neg_a=*/false, a.v, /*neg_b=*/false, b.v,
        /*c_mod=*/(short)0, c, /*reuse_a=*/false, /*reuse_b=*/false);
}

// Problem constants
#define BATCH 4
#define SEQ   2048
#define DMODEL 1024
#define NHEAD 16
#define DHEAD 64
#define MROWS (BATCH * SEQ)   // 8192

// ---------------------------------------------------------------------------
// fp32 -> bf16 elementwise convert (vector x4)
// ---------------------------------------------------------------------------
__global__ void cvt_f32_bf16_x4(const float* __restrict__ src,
                                bf16* __restrict__ dst, int n4) {
    int i = blockIdx.x * blockDim.x + threadIdx.x;
    if (i >= n4) return;
    f32x4 v = *(const f32x4*)(src + 4 * (size_t)i);
    bf16x4 o;
    o.x = (bf16)v.x; o.y = (bf16)v.y; o.z = (bf16)v.z; o.w = (bf16)v.w;
    *(bf16x4*)(dst + 4 * (size_t)i) = o;
}

// ---------------------------------------------------------------------------
// 1024x1024 weight: Wt[n][k] = (bf16)W[k][n]
// ---------------------------------------------------------------------------
__global__ void transpose_cvt_1024(const float* __restrict__ W,
                                   bf16* __restrict__ Wt) {
    int t = blockIdx.x * blockDim.x + threadIdx.x;   // 0..1048575
    int n = t >> 10;
    int k = t & 1023;
    Wt[t] = (bf16)W[(size_t)k * 1024 + n];
}

// ---------------------------------------------------------------------------
// Tiled bf16 WMMA GEMM:  C[M,N] = A[M,K] * Bt[N,K]^T
// Block tile 128x128, 8 waves (2x4), wave tile 64x32, K-step 32.
// Tile staging uses GLOBAL_LOAD_ASYNC_TO_LDS_B128 (ASYNCcnt) with 2-deep
// LDS double buffering: next tile's async copies overlap this tile's WMMAs.
// mode 0: C fp32 row-major (final output)
// mode 1: C bf16 [B,H,S,DH]   (Q, scale folds 1/sqrt(DH)*log2e)
// mode 2: C bf16 [B,H,S,DH]   (K)
// mode 3: C bf16 [B,H,DH,S]   (V, transposed for attention)
// ---------------------------------------------------------------------------
#define GTM 128
#define GTN 128
#define GTK 32
#define LDP 40   // padded LDS row length (bf16 elements)

__global__ __launch_bounds__(256) void gemm_bf16_wmma(
    const bf16* __restrict__ A, const bf16* __restrict__ Bt,
    void* __restrict__ C, int M, int N, int K, int mode, float scale)
{
    __shared__ bf16 As[2 * GTM * LDP];
    __shared__ bf16 Bs[2 * GTN * LDP];

    const int tid  = threadIdx.x;
    const int lane = tid & 31;
    const int wave = tid >> 5;
    const int wm   = wave >> 2;          // 0..1
    const int wn   = wave & 3;           // 0..3
    const int lm   = lane & 15;
    const int ch   = lane >> 4;          // lane half
    const int kb   = ch << 3;            // K sub-offset for fragments
    const int m0   = blockIdx.x * GTM;
    const int n0   = blockIdx.y * GTN;

    // This thread's two staging chunks (8 bf16 = 16 B each, per tile)
    const int c0   = tid << 1;           // chunk ids c0, c0+1
    const int row0 = c0 >> 2;
    const int ko0  = (c0 & 3) << 3;
    const int row1 = (c0 + 1) >> 2;
    const int ko1  = ((c0 + 1) & 3) << 3;

    // Async copy of one K-tile into LDS buffer `buf` (no VGPR staging).
    auto prefetch = [&](int k0, int buf) {
        unsigned la0 = (unsigned)(size_t)&As[buf * (GTM * LDP) + row0 * LDP + ko0];
        unsigned lb0 = (unsigned)(size_t)&Bs[buf * (GTN * LDP) + row0 * LDP + ko0];
        unsigned la1 = (unsigned)(size_t)&As[buf * (GTM * LDP) + row1 * LDP + ko1];
        unsigned lb1 = (unsigned)(size_t)&Bs[buf * (GTN * LDP) + row1 * LDP + ko1];
        const bf16* ga0 = &A [(size_t)(m0 + row0) * K + (k0 + ko0)];
        const bf16* gb0 = &Bt[(size_t)(n0 + row0) * K + (k0 + ko0)];
        const bf16* ga1 = &A [(size_t)(m0 + row1) * K + (k0 + ko1)];
        const bf16* gb1 = &Bt[(size_t)(n0 + row1) * K + (k0 + ko1)];
        asm volatile("global_load_async_to_lds_b128 %0, %1, off"
                     :: "v"(la0), "v"(ga0) : "memory");
        asm volatile("global_load_async_to_lds_b128 %0, %1, off"
                     :: "v"(lb0), "v"(gb0) : "memory");
        asm volatile("global_load_async_to_lds_b128 %0, %1, off"
                     :: "v"(la1), "v"(ga1) : "memory");
        asm volatile("global_load_async_to_lds_b128 %0, %1, off"
                     :: "v"(lb1), "v"(gb1) : "memory");
    };

    f32x8 acc[4][2] = {};

    prefetch(0, 0);
    asm volatile("s_wait_asynccnt 0x0" ::: "memory");
    __syncthreads();

    int buf = 0;
    for (int k0 = 0; k0 < K; k0 += GTK) {
        if (k0 + GTK < K) prefetch(k0 + GTK, buf ^ 1);   // overlap with WMMAs

        const bf16* asbase = &As[buf * (GTM * LDP)];
        const bf16* bsbase = &Bs[buf * (GTN * LDP)];

        Frag bfr[2];
        #pragma unroll
        for (int nt = 0; nt < 2; ++nt) {
            const bf16* p = &bsbase[(wn * 32 + nt * 16 + lm) * LDP + kb];
            bfr[nt].h.lo = *(const bf16x8*)p;
            bfr[nt].h.hi = *(const bf16x8*)(p + 16);
        }
        #pragma unroll
        for (int mt = 0; mt < 4; ++mt) {
            Frag afr;
            const bf16* p = &asbase[(wm * 64 + mt * 16 + lm) * LDP + kb];
            afr.h.lo = *(const bf16x8*)p;
            afr.h.hi = *(const bf16x8*)(p + 16);
            #pragma unroll
            for (int nt = 0; nt < 2; ++nt)
                acc[mt][nt] = wmma_bf16(afr, bfr[nt], acc[mt][nt]);
        }

        asm volatile("s_wait_asynccnt 0x0" ::: "memory");
        __syncthreads();   // all reads of `buf` done + next tile landed
        buf ^= 1;
    }

    // Epilogue: C layout — VGPR r: row = r + 8*lanehalf, col = lane&15
    #pragma unroll
    for (int mt = 0; mt < 4; ++mt)
        #pragma unroll
        for (int nt = 0; nt < 2; ++nt)
            #pragma unroll
            for (int r = 0; r < 8; ++r) {
                int m = m0 + wm * 64 + mt * 16 + r + ch * 8;
                int n = n0 + wn * 32 + nt * 16 + lm;
                float v = acc[mt][nt][r] * scale;
                if (mode == 0) {
                    ((float*)C)[(size_t)m * N + n] = v;
                } else {
                    int b = m >> 11, s = m & 2047;   // S = 2048
                    int h = n >> 6,  d = n & 63;     // DH = 64
                    bf16* O = (bf16*)C;
                    size_t idx = (mode == 3)
                        ? ((size_t)((b * NHEAD + h) * DHEAD + d) * SEQ + s)
                        : ((size_t)((b * NHEAD + h) * SEQ + s) * DHEAD + d);
                    O[idx] = (bf16)v;
                }
            }
}

// ---------------------------------------------------------------------------
// Flash attention, causal. One wave handles one (b, h, 16-query tile).
// Q,K are [B,H,S,DH] bf16 (Q pre-scaled by 1/8*log2e); V is [B,H,DH,S] bf16.
// Computes S^T = K * Q^T so the softmaxed tile feeds O^T = V^T * P^T as the
// B operand directly in-register (no LDS transpose). Output -> [B,S,H*DH] bf16.
// ---------------------------------------------------------------------------
__global__ __launch_bounds__(128) void attn_flash_wmma(
    const bf16* __restrict__ Qb, const bf16* __restrict__ Kb,
    const bf16* __restrict__ Vb, bf16* __restrict__ Ob)
{
    const int lane = threadIdx.x & 31;
    const int wave = threadIdx.x >> 5;
    const int task = blockIdx.x * 4 + wave;    // 0..8191
    const int qt = task & 127;                 // query tile within (b,h)
    const int bh = task >> 7;                  // b*16+h
    const int q0 = qt * 16;

    const bf16* Qp = Qb + ((size_t)bh * SEQ + q0) * DHEAD;
    const bf16* Kp = Kb + (size_t)bh * SEQ * DHEAD;
    const bf16* Vp = Vb + (size_t)bh * DHEAD * SEQ;

    const int lm = lane & 15;
    const int ch = lane >> 4;
    const int kb = ch << 3;

    // Q as B-operand fragments (lane = query column), two K(d)-steps
    Frag qf0, qf1;
    {
        const bf16* qrow = Qp + (size_t)lm * DHEAD;
        qf0.h.lo = *(const bf16x8*)(qrow + kb);
        qf0.h.hi = *(const bf16x8*)(qrow + kb + 16);
        qf1.h.lo = *(const bf16x8*)(qrow + kb + 32);
        qf1.h.hi = *(const bf16x8*)(qrow + kb + 48);
    }

    f32x8 acc[4] = {};              // O^T tiles: d-rows 0..63, queries in lanes
    float m_i = -1e30f, l_i = 0.0f;
    const int qcol = q0 + lm;
    const int kt_end = (q0 + 15) >> 5;   // inclusive, 32-key tiles

    for (int kt = 0; kt <= kt_end; ++kt) {
        const int k0 = kt * 32;
        // ---- S^T (keys x queries), two 16-key M-tiles, reduce over d=64 ----
        f32x8 st0 = {}, st1 = {};
        {
            const bf16* krow = Kp + (size_t)(k0 + lm) * DHEAD;
            Frag a0, a1;
            a0.h.lo = *(const bf16x8*)(krow + kb);
            a0.h.hi = *(const bf16x8*)(krow + kb + 16);
            a1.h.lo = *(const bf16x8*)(krow + kb + 32);
            a1.h.hi = *(const bf16x8*)(krow + kb + 48);
            st0 = wmma_bf16(a0, qf0, st0);
            st0 = wmma_bf16(a1, qf1, st0);
        }
        {
            const bf16* krow = Kp + (size_t)(k0 + 16 + lm) * DHEAD;
            Frag a0, a1;
            a0.h.lo = *(const bf16x8*)(krow + kb);
            a0.h.hi = *(const bf16x8*)(krow + kb + 16);
            a1.h.lo = *(const bf16x8*)(krow + kb + 32);
            a1.h.hi = *(const bf16x8*)(krow + kb + 48);
            st1 = wmma_bf16(a0, qf0, st1);
            st1 = wmma_bf16(a1, qf1, st1);
        }
        // ---- causal mask + per-query running max (mostly in-lane) ----
        float tmax = -1e30f;
        #pragma unroll
        for (int r = 0; r < 8; ++r) {
            int key0 = k0 + r + ch * 8;
            int key1 = key0 + 16;
            float s0 = (key0 <= qcol) ? st0[r] : -1e30f;
            float s1 = (key1 <= qcol) ? st1[r] : -1e30f;
            st0[r] = s0; st1[r] = s1;
            tmax = fmaxf(tmax, fmaxf(s0, s1));
        }
        tmax = fmaxf(tmax, __shfl_xor(tmax, 16, 32));
        float m_new = fmaxf(m_i, tmax);
        float rescale = __builtin_amdgcn_exp2f(m_i - m_new);

        // ---- exponentiate (log2 domain; scale folded into Q) & pack bf16 ----
        Frag pf;
        float psum = 0.0f;
        #pragma unroll
        for (int r = 0; r < 8; ++r) {
            float p0 = __builtin_amdgcn_exp2f(st0[r] - m_new);
            float p1 = __builtin_amdgcn_exp2f(st1[r] - m_new);
            psum += p0 + p1;
            pf.h.lo[r] = (bf16)p0;   // keys kb..kb+7
            pf.h.hi[r] = (bf16)p1;   // keys kb+16..kb+23
        }
        psum += __shfl_xor(psum, 16, 32);
        l_i = l_i * rescale + psum;
        m_i = m_new;

        // ---- O^T += V^T * P^T ; rescale is per-lane (per-query) ----
        #pragma unroll
        for (int mt = 0; mt < 4; ++mt) {
            #pragma unroll
            for (int r = 0; r < 8; ++r) acc[mt][r] *= rescale;
            const bf16* vrow = Vp + (size_t)(mt * 16 + lm) * SEQ + k0;
            Frag va;
            va.h.lo = *(const bf16x8*)(vrow + kb);
            va.h.hi = *(const bf16x8*)(vrow + kb + 16);
            acc[mt] = wmma_bf16(va, pf, acc[mt]);
        }
    }

    // ---- normalize and store to [B,S,H*DH] bf16 for the output GEMM ----
    const float inv = 1.0f / l_i;
    const int b = bh >> 4, h = bh & 15;
    #pragma unroll
    for (int mt = 0; mt < 4; ++mt)
        #pragma unroll
        for (int r = 0; r < 8; ++r) {
            int d = mt * 16 + r + ch * 8;
            int q = q0 + lm;
            Ob[(size_t)(b * SEQ + q) * DMODEL + h * DHEAD + d] =
                (bf16)(acc[mt][r] * inv);
        }
}

// ---------------------------------------------------------------------------
// Host launch
// ---------------------------------------------------------------------------
extern "C" void kernel_launch(void* const* d_in, const int* in_sizes, int n_in,
                              void* d_out, int out_size, void* d_ws, size_t ws_size,
                              hipStream_t stream) {
    (void)in_sizes; (void)n_in; (void)out_size; (void)ws_size;
    const float* x  = (const float*)d_in[0];
    const float* Wq = (const float*)d_in[1];
    const float* Wk = (const float*)d_in[2];
    const float* Wv = (const float*)d_in[3];
    const float* Wo = (const float*)d_in[4];

    char* ws = (char*)d_ws;
    const size_t XB   = (size_t)MROWS * DMODEL * sizeof(bf16);   // 16 MB
    const size_t WB   = (size_t)DMODEL * DMODEL * sizeof(bf16);  //  2 MB
    bf16* xb    = (bf16*)(ws);
    bf16* Wqt   = (bf16*)(ws + XB);
    bf16* Wkt   = (bf16*)(ws + XB + WB);
    bf16* Wvt   = (bf16*)(ws + XB + 2 * WB);
    bf16* Wot   = (bf16*)(ws + XB + 3 * WB);
    bf16* Qb    = (bf16*)(ws + XB + 4 * WB);
    bf16* Kb    = (bf16*)(ws + 2 * XB + 4 * WB);
    bf16* Vb    = (bf16*)(ws + 3 * XB + 4 * WB);
    bf16* attnb = (bf16*)(ws + 4 * XB + 4 * WB);

    // 1) precision conversion / weight transposes
    {
        int n4 = (MROWS * DMODEL) / 4;                 // 2,097,152
        cvt_f32_bf16_x4<<<n4 / 256, 256, 0, stream>>>(x, xb, n4);
        int nt = DMODEL * DMODEL;                      // 1,048,576
        transpose_cvt_1024<<<nt / 256, 256, 0, stream>>>(Wq, Wqt);
        transpose_cvt_1024<<<nt / 256, 256, 0, stream>>>(Wk, Wkt);
        transpose_cvt_1024<<<nt / 256, 256, 0, stream>>>(Wv, Wvt);
        transpose_cvt_1024<<<nt / 256, 256, 0, stream>>>(Wo, Wot);
    }

    // 2) QKV projections (Q gets 1/sqrt(DH)*log2e folded in for exp2 softmax)
    dim3 gg(MROWS / GTM, DMODEL / GTN);   // 64 x 8
    const float QSCALE = 0.125f * 1.4426950408889634f;
    gemm_bf16_wmma<<<gg, 256, 0, stream>>>(xb, Wqt, Qb, MROWS, DMODEL, DMODEL, 1, QSCALE);
    gemm_bf16_wmma<<<gg, 256, 0, stream>>>(xb, Wkt, Kb, MROWS, DMODEL, DMODEL, 2, 1.0f);
    gemm_bf16_wmma<<<gg, 256, 0, stream>>>(xb, Wvt, Vb, MROWS, DMODEL, DMODEL, 3, 1.0f);

    // 3) causal flash attention: 8192 wave-tasks, 4 waves/block
    attn_flash_wmma<<<(BATCH * NHEAD * (SEQ / 16)) / 4, 128, 0, stream>>>(Qb, Kb, Vb, attnb);

    // 4) output projection -> fp32 d_out
    gemm_bf16_wmma<<<gg, 256, 0, stream>>>(attnb, Wot, d_out, MROWS, DMODEL, DMODEL, 0, 1.0f);
}
